// MultiHeadAttention_13967233646855
// MI455X (gfx1250) — matrix operand order repaired and checked by reading, not compiled, not run on hardware
//
#include <hip/hip_runtime.h>
#include <hip/hip_bf16.h>
#include <stdint.h>

// Problem constants (match reference)
#define BB 2
#define SS 2048
#define DD 1024
#define HH 16
#define DH 64
#define NROW (BB * SS)          // 4096
#define LN_EPS 1e-5f

typedef __attribute__((ext_vector_type(16))) __bf16 v16bf;
typedef __attribute__((ext_vector_type(8)))  __bf16 v8bf;
typedef __attribute__((ext_vector_type(8)))  float  v8f;

union BFrag { v16bf v; v8bf h[2]; };

__device__ __forceinline__ __bf16 f2bf(float f) {
  unsigned u = __builtin_bit_cast(unsigned, f);
  u += 0x7FFFu + ((u >> 16) & 1u);           // round-to-nearest-even
  unsigned short hb = (unsigned short)(u >> 16);
  return __builtin_bit_cast(__bf16, hb);
}

__device__ __forceinline__ v8f wmma_bf16(const BFrag& a, const BFrag& b, v8f c) {
  return __builtin_amdgcn_wmma_f32_16x16x32_bf16(false, a.v, false, b.v,
                                                 (short)0, c, false, false);
}

__device__ __forceinline__ float rowmax16(float v) {
  v = fmaxf(v, __shfl_xor(v, 1, 32));
  v = fmaxf(v, __shfl_xor(v, 2, 32));
  v = fmaxf(v, __shfl_xor(v, 4, 32));
  v = fmaxf(v, __shfl_xor(v, 8, 32));
  return v;
}
__device__ __forceinline__ float rowsum16(float v) {
  v += __shfl_xor(v, 1, 32);
  v += __shfl_xor(v, 2, 32);
  v += __shfl_xor(v, 4, 32);
  v += __shfl_xor(v, 8, 32);
  return v;
}

// ---- CDNA5 async global->LDS copy path (ASYNCcnt), guarded so the file
// ---- still compiles if the builtins are not declared by this toolchain.
#if __has_builtin(__builtin_amdgcn_global_load_async_to_lds_b128) && \
    __has_builtin(__builtin_amdgcn_s_wait_asynccnt)
#define USE_ASYNC_LDS 1
#else
#define USE_ASYNC_LDS 0
#endif

#if USE_ASYNC_LDS
typedef int v4i_b __attribute__((vector_size(16)));
typedef __attribute__((address_space(1))) v4i_b* gv4i_p;   // global src
typedef __attribute__((address_space(3))) v4i_b* lv4i_p;   // LDS dst
#endif

__device__ __forceinline__ void cp16_async(const __bf16* gsrc, __bf16* ldst) {
#if USE_ASYNC_LDS
  __builtin_amdgcn_global_load_async_to_lds_b128(
      (gv4i_p)gsrc, (lv4i_p)ldst, 0, 0);
#else
  *(v8bf*)ldst = *(const v8bf*)gsrc;
#endif
}
__device__ __forceinline__ void async_join() {
#if USE_ASYNC_LDS
  __builtin_amdgcn_s_wait_asynccnt(0);
#endif
}

// ---------------------------------------------------------------------------
// Kernel 0a: fp32 -> bf16 conversion of q/k/v (one pass, so GEMM staging
// becomes raw 16-byte copies eligible for global_load_async_to_lds_b128).
// ---------------------------------------------------------------------------
__global__ __launch_bounds__(256) void x_to_bf16(
    const float* __restrict__ q, const float* __restrict__ k,
    const float* __restrict__ v,
    __bf16* __restrict__ qb, __bf16* __restrict__ kb, __bf16* __restrict__ vb)
{
  const int z = blockIdx.y;
  const float* X = (z == 0) ? q : (z == 1) ? k : v;
  __bf16* Xb     = (z == 0) ? qb : (z == 1) ? kb : vb;
  size_t i = ((size_t)blockIdx.x * 256 + threadIdx.x) * 4;
  const float4 f = *(const float4*)(X + i);
  Xb[i + 0] = f2bf(f.x); Xb[i + 1] = f2bf(f.y);
  Xb[i + 2] = f2bf(f.z); Xb[i + 3] = f2bf(f.w);
}

// ---------------------------------------------------------------------------
// Kernel 0b: convert + transpose weights: Wt[n][k] = bf16(W[k][n]).
// 32x32 LDS tile transpose, z selects Wq/Wk/Wv/Wo.
// ---------------------------------------------------------------------------
__global__ __launch_bounds__(256) void w_transpose_bf16(
    const float* __restrict__ Wq, const float* __restrict__ Wk,
    const float* __restrict__ Wv, const float* __restrict__ Wo,
    __bf16* __restrict__ Wqt, __bf16* __restrict__ Wkt,
    __bf16* __restrict__ Wvt, __bf16* __restrict__ Wot)
{
  const int z = blockIdx.z;
  const float* W = (z == 0) ? Wq : (z == 1) ? Wk : (z == 2) ? Wv : Wo;
  __bf16* Wt     = (z == 0) ? Wqt : (z == 1) ? Wkt : (z == 2) ? Wvt : Wot;

  __shared__ __bf16 t[32][33];
  const int lx = threadIdx.x & 31, ly = threadIdx.x >> 5;   // 32 x 8
  const int k0 = blockIdx.x * 32, n0 = blockIdx.y * 32;
  #pragma unroll
  for (int r = 0; r < 4; ++r)
    t[ly + r * 8][lx] = f2bf(W[(size_t)(k0 + ly + r * 8) * DD + n0 + lx]);
  __syncthreads();
  #pragma unroll
  for (int r = 0; r < 4; ++r)
    Wt[(size_t)(n0 + ly + r * 8) * DD + k0 + lx] = t[lx][ly + r * 8];
}

// ---------------------------------------------------------------------------
// Shared GEMM core: 128(M) x 64(N) block tile, K stepped by 64, 8 waves,
// each wave owns a 32x32 tile (2x2 WMMA 16x16x32_bf16 accumulators).
// A: bf16 [.,1024] row-major starting at row m0.
// Bt: bf16 transposed weights [n][k], starting at row n0.
// Staging is raw 16B chunks via async-to-LDS; next-K prefetch hints issued.
// ---------------------------------------------------------------------------
#define KSTEP 64
#define LPAD  72   // 64 + 8 halves; row stride 144B = 9*16B (keeps 16B align)

__device__ __forceinline__ void gemm_mac_128x64(
    const __bf16* __restrict__ Abase,   // A + m0*DD
    const __bf16* __restrict__ Bbase,   // Bt + n0*DD
    __bf16 (*As)[LPAD], __bf16 (*Bs)[LPAD],
    int tid, v8f acc[2][2])
{
  const int lane = tid & 31, wid = tid >> 5;
  const int waveM = wid & 3, waveN = wid >> 2;
  const int nl = lane & 15, half = lane >> 4;

  for (int kt = 0; kt < DD; kt += KSTEP) {
    // stage A: 128 x 64 bf16 = 1024 x 16B chunks (4 per thread)
    #pragma unroll
    for (int i = 0; i < 4; ++i) {
      int c = tid + 256 * i;
      int row = c >> 3, off = (c & 7) * 8;
      const __bf16* src = Abase + (size_t)row * DD + kt + off;
      cp16_async(src, &As[row][off]);
      if (i == 0 && kt + KSTEP < DD) __builtin_prefetch(src + KSTEP, 0, 0);
    }
    // stage B: 64 x 64 bf16 = 512 x 16B chunks (2 per thread)
    #pragma unroll
    for (int i = 0; i < 2; ++i) {
      int c = tid + 256 * i;
      int row = c >> 3, off = (c & 7) * 8;
      const __bf16* src = Bbase + (size_t)row * DD + kt + off;
      cp16_async(src, &Bs[row][off]);
      if (i == 0 && kt + KSTEP < DD) __builtin_prefetch(src + KSTEP, 0, 0);
    }
    async_join();
    __syncthreads();

    #pragma unroll
    for (int ss = 0; ss < 2; ++ss) {          // two K=32 sub-steps
      BFrag a[2], b[2];
      #pragma unroll
      for (int i = 0; i < 2; ++i) {
        const __bf16* ap = &As[waveM * 32 + i * 16 + nl][ss * 32 + half * 8];
        a[i].h[0] = *(const v8bf*)ap;
        a[i].h[1] = *(const v8bf*)(ap + 16);
      }
      #pragma unroll
      for (int j = 0; j < 2; ++j) {
        const __bf16* bp = &Bs[waveN * 32 + j * 16 + nl][ss * 32 + half * 8];
        b[j].h[0] = *(const v8bf*)bp;
        b[j].h[1] = *(const v8bf*)(bp + 16);
      }
      #pragma unroll
      for (int i = 0; i < 2; ++i)
        #pragma unroll
        for (int j = 0; j < 2; ++j)
          acc[i][j] = wmma_bf16(a[i], b[j], acc[i][j]);
    }
    __syncthreads();
  }
}

// ---------------------------------------------------------------------------
// Kernel 1: QKV projection (z selects X/W/bias/output).
// Q,K stored bf16 [B,H,S,DH]; V stored transposed bf16 [B,H,DH,S].
// ---------------------------------------------------------------------------
__global__ __launch_bounds__(256) void qkv_gemm(
    const __bf16* __restrict__ qb, const __bf16* __restrict__ kb,
    const __bf16* __restrict__ vb,
    const __bf16* __restrict__ Wqt, const __bf16* __restrict__ Wkt,
    const __bf16* __restrict__ Wvt,
    const float* __restrict__ bq, const float* __restrict__ bk,
    const float* __restrict__ bv,
    __bf16* __restrict__ Qd, __bf16* __restrict__ Kd, __bf16* __restrict__ Vt)
{
  const int z = blockIdx.z;
  const __bf16* X    = (z == 0) ? qb  : (z == 1) ? kb  : vb;
  const __bf16* Bt   = (z == 0) ? Wqt : (z == 1) ? Wkt : Wvt;
  const float*  bias = (z == 0) ? bq  : (z == 1) ? bk  : bv;

  __shared__ __bf16 As[128][LPAD];
  __shared__ __bf16 Bs[64][LPAD];

  const int m0 = blockIdx.x * 128;
  const int n0 = blockIdx.y * 64;
  const int tid = threadIdx.x;
  const int lane = tid & 31, wid = tid >> 5;
  const int waveM = wid & 3, waveN = wid >> 2;
  const int nl = lane & 15, half = lane >> 4;

  v8f acc[2][2] = {};
  gemm_mac_128x64(X + (size_t)m0 * DD, Bt + (size_t)n0 * DD, As, Bs, tid, acc);

  #pragma unroll
  for (int i = 0; i < 2; ++i) {
    #pragma unroll
    for (int j = 0; j < 2; ++j) {
      int ng = n0 + waveN * 32 + j * 16 + nl;
      float bv_ = bias[ng];
      int h = ng >> 6, d = ng & 63;
      #pragma unroll
      for (int r = 0; r < 8; ++r) {
        int mg = m0 + waveM * 32 + i * 16 + half * 8 + r;
        int bidx = mg >> 11, s = mg & (SS - 1);
        float val = acc[i][j][r] + bv_;
        if (z == 2) {
          Vt[(((size_t)bidx * HH + h) * DH + d) * SS + s] = f2bf(val);
        } else {
          size_t off = (((size_t)bidx * HH + h) * SS + s) * DH + d;
          if (z == 0) Qd[off] = f2bf(val);
          else        Kd[off] = f2bf(val);
        }
      }
    }
  }
}

// ---------------------------------------------------------------------------
// Kernel 2: flash attention per (b,h). 128 threads = 4 waves, 16 q-rows/wave,
// online softmax over S_k in chunks of 32 keys (8 WMMAs per chunk).
// ---------------------------------------------------------------------------
__global__ __launch_bounds__(128) void attn_kernel(
    const __bf16* __restrict__ Qd, const __bf16* __restrict__ Kd,
    const __bf16* __restrict__ Vt, const float* __restrict__ mask,
    __bf16* __restrict__ Ctx)
{
  __shared__ __bf16 Ps[4][16][32];           // per-wave P restage buffer

  const int lane = threadIdx.x & 31, w = threadIdx.x >> 5;
  const int nl = lane & 15, half = lane >> 4;
  const int bh = blockIdx.y;                 // b*H + h
  const int b = bh >> 4, h = bh & 15;
  const int q0 = blockIdx.x * 64 + w * 16;

  const __bf16* Qb = Qd + (size_t)bh * SS * DH;
  const __bf16* Kb = Kd + (size_t)bh * SS * DH;
  const __bf16* Vb = Vt + (size_t)bh * DH * SS;
  const float*  Mb = mask + (size_t)b * SS * SS;

  // Q A-fragments, register-resident for the whole key loop
  BFrag aq[2];
  #pragma unroll
  for (int ks = 0; ks < 2; ++ks) {
    const __bf16* p = Qb + (size_t)(q0 + nl) * DH + ks * 32 + half * 8;
    aq[ks].h[0] = *(const v8bf*)p;
    aq[ks].h[1] = *(const v8bf*)(p + 16);
  }

  float mrun[8], lrun[8];
  #pragma unroll
  for (int r = 0; r < 8; ++r) { mrun[r] = -1e30f; lrun[r] = 0.f; }
  v8f ctx[4] = {};

  for (int kc = 0; kc < SS; kc += 32) {
    if (kc + 32 < SS) {   // prefetch hints for next chunk (global_prefetch_b8)
      __builtin_prefetch(Kb + (size_t)(kc + 32 + nl) * DH, 0, 0);
      __builtin_prefetch(Vb + (size_t)nl * SS + kc + 32, 0, 0);
      __builtin_prefetch(Mb + (size_t)(q0 + half * 8) * SS + kc + 32, 0, 0);
    }

    // ---- scores: two 16x16 tiles over key columns [kc, kc+32)
    v8f s0 = {}, s1 = {};
    #pragma unroll
    for (int ks = 0; ks < 2; ++ks) {
      BFrag bk0, bk1;
      const __bf16* p0 = Kb + (size_t)(kc + nl) * DH + ks * 32 + half * 8;
      bk0.h[0] = *(const v8bf*)p0; bk0.h[1] = *(const v8bf*)(p0 + 16);
      const __bf16* p1 = Kb + (size_t)(kc + 16 + nl) * DH + ks * 32 + half * 8;
      bk1.h[0] = *(const v8bf*)p1; bk1.h[1] = *(const v8bf*)(p1 + 16);
      s0 = wmma_bf16(aq[ks], bk0, s0);
      s1 = wmma_bf16(aq[ks], bk1, s1);
    }

    // ---- scale + multiplicative mask
    float p0v[8], p1v[8];
    #pragma unroll
    for (int r = 0; r < 8; ++r) {
      int m = half * 8 + r;
      const float* mrow = Mb + (size_t)(q0 + m) * SS + kc;
      p0v[r] = s0[r] * 0.125f * mrow[nl];
      p1v[r] = s1[r] * 0.125f * mrow[16 + nl];
    }

    // ---- online softmax (rows stay inside one lane-half)
    float corr[8];
    #pragma unroll
    for (int r = 0; r < 8; ++r) {
      float cmax = rowmax16(fmaxf(p0v[r], p1v[r]));
      float mnew = fmaxf(mrun[r], cmax);
      corr[r] = __expf(mrun[r] - mnew);
      p0v[r] = __expf(p0v[r] - mnew);
      p1v[r] = __expf(p1v[r] - mnew);
      lrun[r] = lrun[r] * corr[r] + rowsum16(p0v[r] + p1v[r]);
      mrun[r] = mnew;
    }
    #pragma unroll
    for (int dt = 0; dt < 4; ++dt)
      #pragma unroll
      for (int r = 0; r < 8; ++r) ctx[dt][r] *= corr[r];

    // ---- restage P (C layout) into an A-fragment via LDS
    #pragma unroll
    for (int r = 0; r < 8; ++r) {
      int m = half * 8 + r;
      Ps[w][m][nl]      = f2bf(p0v[r]);
      Ps[w][m][16 + nl] = f2bf(p1v[r]);
    }
    __syncthreads();
    BFrag ap;
    {
      const __bf16* pp = &Ps[w][nl][half * 8];
      ap.h[0] = *(const v8bf*)pp;
      ap.h[1] = *(const v8bf*)(pp + 16);
    }
    __syncthreads();

    // ---- ctx += P(16x32) @ V(32x64): 4 WMMAs against transposed V
    #pragma unroll
    for (int dt = 0; dt < 4; ++dt) {
      BFrag bvf;
      const __bf16* vp = Vb + (size_t)(dt * 16 + nl) * SS + kc + half * 8;
      bvf.h[0] = *(const v8bf*)vp; bvf.h[1] = *(const v8bf*)(vp + 16);
      ctx[dt] = wmma_bf16(ap, bvf, ctx[dt]);
    }
  }

  // ---- normalize and store ctx bf16 as [B*S, H*DH]
  #pragma unroll
  for (int r = 0; r < 8; ++r) {
    int m = half * 8 + r;
    float inv = 1.f / lrun[r];
    size_t rowg = ((size_t)b * SS + q0 + m) * (HH * DH) + (size_t)h * DH;
    #pragma unroll
    for (int dt = 0; dt < 4; ++dt)
      Ctx[rowg + dt * 16 + nl] = f2bf(ctx[dt][r] * inv);
  }
}

// ---------------------------------------------------------------------------
// Kernel 3: out = ctx @ Wo + bo + q (residual), fp32 to scratch Y.
// ---------------------------------------------------------------------------
__global__ __launch_bounds__(256) void out_gemm(
    const __bf16* __restrict__ Ctx, const __bf16* __restrict__ Wot,
    const float* __restrict__ bo, const float* __restrict__ qin,
    float* __restrict__ Y)
{
  __shared__ __bf16 As[128][LPAD];
  __shared__ __bf16 Bs[64][LPAD];

  const int m0 = blockIdx.x * 128;
  const int n0 = blockIdx.y * 64;
  const int tid = threadIdx.x;
  const int lane = tid & 31, wid = tid >> 5;
  const int waveM = wid & 3, waveN = wid >> 2;
  const int nl = lane & 15, half = lane >> 4;

  v8f acc[2][2] = {};
  gemm_mac_128x64(Ctx + (size_t)m0 * DD, Wot + (size_t)n0 * DD, As, Bs, tid, acc);

  #pragma unroll
  for (int i = 0; i < 2; ++i) {
    #pragma unroll
    for (int j = 0; j < 2; ++j) {
      int ng = n0 + waveN * 32 + j * 16 + nl;
      float bv_ = bo[ng];
      #pragma unroll
      for (int r = 0; r < 8; ++r) {
        int mg = m0 + waveM * 32 + i * 16 + half * 8 + r;
        size_t off = (size_t)mg * DD + ng;
        Y[off] = acc[i][j][r] + bv_ + qin[off];
      }
    }
  }
}

// ---------------------------------------------------------------------------
// Kernel 4: row LayerNorm over D=1024. One 256-thread block per row.
// ---------------------------------------------------------------------------
__global__ __launch_bounds__(256) void layernorm_kernel(
    const float* __restrict__ Y, const float* __restrict__ gamma,
    const float* __restrict__ beta, float* __restrict__ out)
{
  __shared__ float red[2][8];
  const int row = blockIdx.x;
  const float* y = Y + (size_t)row * DD;

  float s = 0.f, ss = 0.f, vals[4];
  #pragma unroll
  for (int i = 0; i < 4; ++i) {
    float v = y[threadIdx.x + i * 256];
    vals[i] = v; s += v; ss += v * v;
  }
  #pragma unroll
  for (int off = 1; off < 32; off <<= 1) {
    s  += __shfl_xor(s, off, 32);
    ss += __shfl_xor(ss, off, 32);
  }
  const int lane = threadIdx.x & 31, w = threadIdx.x >> 5;
  if (lane == 0) { red[0][w] = s; red[1][w] = ss; }
  __syncthreads();
  if (w == 0) {
    float a  = (lane < 8) ? red[0][lane] : 0.f;
    float b2 = (lane < 8) ? red[1][lane] : 0.f;
    #pragma unroll
    for (int off = 1; off < 8; off <<= 1) {
      a  += __shfl_xor(a, off, 32);
      b2 += __shfl_xor(b2, off, 32);
    }
    if (lane == 0) { red[0][0] = a; red[1][0] = b2; }
  }
  __syncthreads();
  const float mu   = red[0][0] * (1.f / DD);
  const float var  = red[1][0] * (1.f / DD) - mu * mu;
  const float rinv = rsqrtf(var + LN_EPS);
  #pragma unroll
  for (int i = 0; i < 4; ++i) {
    int c = threadIdx.x + i * 256;
    out[(size_t)row * DD + c] = (vals[i] - mu) * rinv * gamma[c] + beta[c];
  }
}

// ---------------------------------------------------------------------------
extern "C" void kernel_launch(void* const* d_in, const int* in_sizes, int n_in,
                              void* d_out, int out_size, void* d_ws, size_t ws_size,
                              hipStream_t stream) {
  (void)in_sizes; (void)n_in; (void)out_size; (void)ws_size;

  const float* q     = (const float*)d_in[0];
  const float* k     = (const float*)d_in[1];
  const float* v     = (const float*)d_in[2];
  const float* mask  = (const float*)d_in[3];
  const float* Wq    = (const float*)d_in[4];
  const float* bq    = (const float*)d_in[5];
  const float* Wk    = (const float*)d_in[6];
  const float* bk    = (const float*)d_in[7];
  const float* Wv    = (const float*)d_in[8];
  const float* bv    = (const float*)d_in[9];
  const float* Wo    = (const float*)d_in[10];
  const float* bo    = (const float*)d_in[11];
  const float* gamma = (const float*)d_in[12];
  const float* beta  = (const float*)d_in[13];

  // Workspace carve-up (~80 MB)
  const size_t MN = (size_t)NROW * DD;                 // 4096*1024
  const size_t WN = (size_t)DD * DD;                   // 1024*1024
  uintptr_t p = ((uintptr_t)d_ws + 255) & ~(uintptr_t)255;
  __bf16* Qd  = (__bf16*)p;  p += MN * sizeof(__bf16);
  __bf16* Kd  = (__bf16*)p;  p += MN * sizeof(__bf16);
  __bf16* Vt  = (__bf16*)p;  p += MN * sizeof(__bf16);
  __bf16* Ctx = (__bf16*)p;  p += MN * sizeof(__bf16);
  __bf16* qb  = (__bf16*)p;  p += MN * sizeof(__bf16);
  __bf16* kb  = (__bf16*)p;  p += MN * sizeof(__bf16);
  __bf16* vb  = (__bf16*)p;  p += MN * sizeof(__bf16);
  __bf16* Wqt = (__bf16*)p;  p += WN * sizeof(__bf16);
  __bf16* Wkt = (__bf16*)p;  p += WN * sizeof(__bf16);
  __bf16* Wvt = (__bf16*)p;  p += WN * sizeof(__bf16);
  __bf16* Wot = (__bf16*)p;  p += WN * sizeof(__bf16);
  float*  Y   = (float*)p;

  dim3 g0a((unsigned)(MN / 4 / 256), 3);
  x_to_bf16<<<g0a, 256, 0, stream>>>(q, k, v, qb, kb, vb);

  dim3 g0b(DD / 32, DD / 32, 4);
  w_transpose_bf16<<<g0b, 256, 0, stream>>>(Wq, Wk, Wv, Wo, Wqt, Wkt, Wvt, Wot);

  dim3 g1(NROW / 128, DD / 64, 3);
  qkv_gemm<<<g1, 256, 0, stream>>>(qb, kb, vb, Wqt, Wkt, Wvt,
                                   bq, bk, bv, Qd, Kd, Vt);

  dim3 g2(SS / 64, BB * HH);
  attn_kernel<<<g2, 128, 0, stream>>>(Qd, Kd, Vt, mask, Ctx);

  dim3 g3(NROW / 128, DD / 64);
  out_gemm<<<g3, 256, 0, stream>>>(Ctx, Wot, bo, q, Y);

  layernorm_kernel<<<NROW, 256, 0, stream>>>(Y, gamma, beta, (float*)d_out);
}
